// GraphDiffusion_280
// MI455X (gfx1250) — matrix operand order repaired and checked by reading, not compiled, not run on hardware
//
#include <hip/hip_runtime.h>
#include <hip/hip_bf16.h>

typedef __attribute__((ext_vector_type(16))) _Float16 v16h;
typedef __attribute__((ext_vector_type(8)))  _Float16 v8h;
typedef __attribute__((ext_vector_type(8)))  float    v8f;

namespace {
constexpr int N    = 50000;
constexpr int E    = 1600000;
constexpr int H    = 64;
constexpr int P    = 5;
constexpr int L    = 3;
constexpr int NH   = N * H;          // 3,200,000
constexpr int NBLK = 512;            // stats partial blocks
constexpr int CHUNK = (N + NBLK - 1) / NBLK;  // 98

// workspace layout (float units)
constexpr size_t OFF_DEG  = 0;                        // N
constexpr size_t OFF_NORM = 50176;                    // E
constexpr size_t OFF_H    = OFF_NORM + (size_t)E;     // NH
constexpr size_t OFF_R    = OFF_H + (size_t)NH;       // NH
constexpr size_t OFF_TA   = OFF_R + (size_t)NH;       // NH (reused as h_f16)
constexpr size_t OFF_TB   = OFF_TA + (size_t)NH;      // NH (reused for W frags)
constexpr size_t OFF_PART = OFF_TB + (size_t)NH;      // NBLK*(H+1)
constexpr size_t OFF_COEF = OFF_PART + (size_t)NBLK * (H + 1); // 8
}

// ---------------- utility ----------------
__global__ void k_zero(float* __restrict__ p, int n) {
  int i = blockIdx.x * blockDim.x + threadIdx.x;
  if (i < n) p[i] = 0.f;
}
__global__ void k_copy(const float* __restrict__ s, float* __restrict__ d) {
  int i = blockIdx.x * blockDim.x + threadIdx.x;
  d[i] = s[i];
}

// ---------------- normalization ----------------
__global__ void k_deg(const int* __restrict__ ei, const float* __restrict__ w,
                      float* __restrict__ deg) {
  int e = blockIdx.x * blockDim.x + threadIdx.x;
  if (e < E) atomicAdd(&deg[ei[E + e]], w[e]);
}
__global__ void k_dis(float* __restrict__ deg) {
  int i = blockIdx.x * blockDim.x + threadIdx.x;
  if (i < N) {
    float d = fminf(rsqrtf(deg[i]), 1e6f);   // min(inf,1e6)=1e6, matches ref
    deg[i] = d;
  }
}
__global__ void k_norm(const int* __restrict__ ei, const float* __restrict__ w,
                       const float* __restrict__ dis, float* __restrict__ nrm) {
  int e = blockIdx.x * blockDim.x + threadIdx.x;
  if (e < E) nrm[e] = dis[ei[e]] * w[e] * dis[ei[E + e]];
}

// ---------------- input projection: h = x @ W_in.T + b_in ----------------
__global__ void k_proj(const float* __restrict__ x, const float* __restrict__ Wi,
                       const float* __restrict__ bi, float* __restrict__ h) {
  int i = blockIdx.x * 256 + threadIdx.x;     // i < NH
  int n = i >> 6, j = i & 63;
  float acc = bi[j];
#pragma unroll
  for (int f = 0; f < 8; ++f) acc += x[n * 8 + f] * Wi[j * 8 + f];
  h[i] = acc;
}

// ---------------- stats: per-feature sums + sum of squares ----------------
__global__ void k_stats(const float* __restrict__ h, float* __restrict__ part) {
  __shared__ float qs[64];
  int j = threadIdx.x;
  int b = blockIdx.x;
  int n0 = b * CHUNK;
  int n1 = n0 + CHUNK; if (n1 > N) n1 = N;
  float s = 0.f, q = 0.f;
  for (int n = n0; n < n1; ++n) { float v = h[n * 64 + j]; s += v; q += v * v; }
  part[b * 65 + j] = s;
  qs[j] = q;
  __syncthreads();
  if (j == 0) { float t = 0.f; for (int k = 0; k < 64; ++k) t += qs[k]; part[b * 65 + 64] = t; }
}

// ---------------- coeff MLP (single block, 64 threads) ----------------
__global__ void k_coeffs(const float* __restrict__ part,
                         const float* __restrict__ cW1, const float* __restrict__ cb1,
                         const float* __restrict__ cW2, const float* __restrict__ cb2,
                         int l, float* __restrict__ coef) {
  __shared__ float sj[64];
  __shared__ float ci[68];
  __shared__ float hid[32];
  __shared__ float o[6];
  int j = threadIdx.x;
  float s = 0.f;
  for (int b = 0; b < NBLK; ++b) s += part[b * 65 + j];
  sj[j] = s;
  __syncthreads();
  if (j == 0) {
    float tot = 0.f; for (int k = 0; k < 64; ++k) tot += sj[k];
    float q = 0.f;   for (int b = 0; b < NBLK; ++b) q += part[b * 65 + 64];
    const float cnt = (float)N * (float)H;
    float mean = tot / cnt;
    float var  = (q - tot * tot / cnt) / (cnt - 1.f);  // unbiased (ddof=1)
    ci[64] = mean;
    ci[65] = sqrtf(fmaxf(var, 0.f));
    ci[66] = (float)N;
    ci[67] = (float)E;
  }
  ci[j] = sj[j] / (float)N;          // per-feature mean
  __syncthreads();
  if (j < 32) {
    const float* wr = cW1 + ((size_t)l * 32 + j) * 68;
    float a = cb1[l * 32 + j];
    for (int k = 0; k < 68; ++k) a += wr[k] * ci[k];
    hid[j] = a > 0.f ? a : 0.f;
  }
  __syncthreads();
  if (j < 6) {
    const float* wr = cW2 + ((size_t)l * 6 + j) * 32;
    float a = cb2[l * 6 + j];
    for (int k = 0; k < 32; ++k) a += wr[k] * hid[k];
    o[j] = a;
  }
  __syncthreads();
  if (j == 0) {
    float m = o[0]; for (int p = 1; p < 6; ++p) m = fmaxf(m, o[p]);
    float ssum = 0.f, ex[6];
    for (int p = 0; p < 6; ++p) { ex[p] = expf(o[p] - m); ssum += ex[p]; }
    for (int p = 0; p < 6; ++p) coef[p] = ex[p] / ssum;
  }
}

// ---------------- diffusion ----------------
__global__ void k_scale(const float* __restrict__ h, float* __restrict__ R,
                        const float* __restrict__ coef) {
  int i = blockIdx.x * 256 + threadIdx.x;
  R[i] = coef[0] * h[i];
}
__global__ void k_axpy(float* __restrict__ R, const float* __restrict__ t,
                       const float* __restrict__ coef, int k) {
  int i = blockIdx.x * 256 + threadIdx.x;
  R[i] += coef[k] * t[i];
}
// SpMM: dst[row] += norm * src[col]   (64 lanes per edge, 4 edges/block)
__global__ void k_spmm(const int* __restrict__ ei, const float* __restrict__ nrm,
                       const float* __restrict__ src, float* __restrict__ dst) {
  int tid = threadIdx.x;
  int e = blockIdx.x * 4 + (tid >> 6);
  int j = tid & 63;
  int r = ei[e];
  int c = ei[E + e];
  float v = nrm[e] * src[c * 64 + j];
  atomicAdd(&dst[r * 64 + j], v);
}

// ---------------- residual + layernorm (in place on h) ----------------
__global__ void k_ln(float* __restrict__ h, const float* __restrict__ R,
                     const float* __restrict__ lns, const float* __restrict__ lnb, int l) {
  __shared__ float sm[256], sq[256];
  int tid = threadIdx.x;
  int g = tid >> 6;
  int j = tid & 63;
  int n = blockIdx.x * 4 + g;
  float v = h[n * 64 + j] + R[n * 64 + j];
  sm[tid] = v; sq[tid] = v * v;
  __syncthreads();
  for (int off = 32; off > 0; off >>= 1) {
    if (j < off) { sm[tid] += sm[tid + off]; sq[tid] += sq[tid + off]; }
    __syncthreads();
  }
  float mean = sm[g * 64] * (1.f / 64.f);
  float var  = sq[g * 64] * (1.f / 64.f) - mean * mean;   // biased, as jnp.var
  float inv = rsqrtf(var + 1e-5f);
  h[n * 64 + j] = (v - mean) * inv * lns[l * 64 + j] + lnb[l * 64 + j];
}

// ---------------- edge predictor prep ----------------
__global__ void k_h2f16(const float* __restrict__ h, _Float16* __restrict__ hf) {
  int i = blockIdx.x * 256 + threadIdx.x;
  hf[i] = (_Float16)h[i];
}
// pack W1(64x128) / W2(32x64) into WMMA A-fragment order
__global__ void k_packw(const float* __restrict__ W1, const float* __restrict__ W2,
                        _Float16* __restrict__ w1f, _Float16* __restrict__ w2f) {
  int i = blockIdx.x * 256 + threadIdx.x;   // 40 blocks * 256 = 10240
  if (i < 8192) {
    int hh = i & 15, lane = (i >> 4) & 31, fk = i >> 9;  // fk = t*4+kk
    int t = fk >> 2, kk = fk & 3;
    int m = t * 16 + (lane & 15);
    int k = kk * 32 + (lane >> 4) * 16 + hh;
    w1f[i] = (_Float16)W1[m * 128 + k];
  } else if (i < 8192 + 2048) {
    int ii = i - 8192;
    int hh = ii & 15, lane = (ii >> 4) & 31, fk = ii >> 9; // fk = t2*2+kk2
    int t2 = fk >> 1, kk2 = fk & 1;
    int m = t2 * 16 + (lane & 15);
    int k = kk2 * 32 + (lane >> 4) * 16 + hh;
    w2f[ii] = (_Float16)W2[m * 64 + k];
  }
}

// ---------------- edge predictor: WMMA f16 MLP (16 edges / wave) ----------------
__global__ __launch_bounds__(256) void
k_edge(const _Float16* __restrict__ hf, const _Float16* __restrict__ w1f,
       const _Float16* __restrict__ w2f, const int* __restrict__ ei,
       const float* __restrict__ b1, const float* __restrict__ b2,
       const float* __restrict__ w3, const float* __restrict__ b3,
       float* __restrict__ preds) {
  __shared__ __align__(32) _Float16 e1s[8 * 16 * 64];  // 16 KB
  __shared__ __align__(16) float    e2s[8 * 16 * 32];  // 16 KB
  const int tid  = threadIdx.x;
  const int wave = tid >> 5;
  const int lane = tid & 31;
  const int ln16 = lane & 15;
  const int hi   = lane >> 4;
  const int tile0 = (blockIdx.x * 8 + wave) * 16;      // E % 128 == 0: no tail
  const int e = tile0 + ln16;
  const int r = ei[e];
  const int c = ei[E + e];

  // B fragments for layer 1 (K=128 over 4 k-steps of 32)
  const _Float16* hr = hf + (size_t)r * H;
  const _Float16* hc = hf + (size_t)c * H;
  v16h bf[4];
  bf[0] = *(const v16h*)(hr + hi * 16);
  bf[1] = *(const v16h*)(hr + 32 + hi * 16);
  bf[2] = *(const v16h*)(hc + hi * 16);
  bf[3] = *(const v16h*)(hc + 32 + hi * 16);

  _Float16* myE1 = e1s + wave * (16 * 64) + ln16 * 64;
#pragma unroll
  for (int t = 0; t < 4; ++t) {
    v8f acc;
    const float* bb = b1 + t * 16 + hi * 8;
#pragma unroll
    for (int q = 0; q < 8; ++q) acc[q] = bb[q];
#pragma unroll
    for (int kk = 0; kk < 4; ++kk) {
      v16h a = *(const v16h*)(w1f + ((t * 4 + kk) * 32 + lane) * 16);
      acc = __builtin_amdgcn_wmma_f32_16x16x32_f16(false, a, false, bf[kk],
                                                   (short)0, acc, false, false);
    }
    v8h pk;
#pragma unroll
    for (int q = 0; q < 8; ++q) { float v = acc[q]; v = v > 0.f ? v : 0.f; pk[q] = (_Float16)v; }
    *(v8h*)(myE1 + t * 16 + hi * 8) = pk;              // ReLU(e1) as f16
  }
  __syncthreads();

  // layer 2: B fragments from LDS (K=64 over 2 k-steps)
  v16h b2f[2];
  b2f[0] = *(const v16h*)(myE1 + hi * 16);
  b2f[1] = *(const v16h*)(myE1 + 32 + hi * 16);
  float* myE2 = e2s + wave * (16 * 32) + ln16 * 32;
#pragma unroll
  for (int t2 = 0; t2 < 2; ++t2) {
    v8f acc;
    const float* bb = b2 + t2 * 16 + hi * 8;
#pragma unroll
    for (int q = 0; q < 8; ++q) acc[q] = bb[q];
#pragma unroll
    for (int kk = 0; kk < 2; ++kk) {
      v16h a = *(const v16h*)(w2f + ((t2 * 2 + kk) * 32 + lane) * 16);
      acc = __builtin_amdgcn_wmma_f32_16x16x32_f16(false, a, false, b2f[kk],
                                                   (short)0, acc, false, false);
    }
#pragma unroll
    for (int q = 0; q < 8; ++q) {
      float v = acc[q]; v = v > 0.f ? v : 0.f;
      myE2[t2 * 16 + hi * 8 + q] = v;                  // ReLU(e2) as f32
    }
  }
  __syncthreads();

  // layer 3: dot(32) + sigmoid, one lane per edge
  if (hi == 0) {
    const float* row32 = e2s + wave * (16 * 32) + ln16 * 32;
    float d = b3[0];
#pragma unroll
    for (int f = 0; f < 32; ++f) d += row32[f] * w3[f];
    preds[e] = 1.f / (1.f + expf(-d));
  }
}

// ---------------- launch ----------------
extern "C" void kernel_launch(void* const* d_in, const int* in_sizes, int n_in,
                              void* d_out, int out_size, void* d_ws, size_t ws_size,
                              hipStream_t stream) {
  (void)in_sizes; (void)n_in; (void)out_size; (void)ws_size;
  const float* x   = (const float*)d_in[0];
  const int*   ei  = (const int*)  d_in[1];
  const float* ew  = (const float*)d_in[2];
  const float* Win = (const float*)d_in[3];
  const float* bin = (const float*)d_in[4];
  const float* cW1 = (const float*)d_in[5];
  const float* cb1 = (const float*)d_in[6];
  const float* cW2 = (const float*)d_in[7];
  const float* cb2 = (const float*)d_in[8];
  const float* lns = (const float*)d_in[9];
  const float* lnb = (const float*)d_in[10];
  const float* eW1 = (const float*)d_in[11];
  const float* eb1 = (const float*)d_in[12];
  const float* eW2 = (const float*)d_in[13];
  const float* eb2 = (const float*)d_in[14];
  const float* eW3 = (const float*)d_in[15];
  const float* eb3 = (const float*)d_in[16];

  float* w    = (float*)d_ws;
  float* deg  = w + OFF_DEG;
  float* nrm  = w + OFF_NORM;
  float* h    = w + OFF_H;
  float* R    = w + OFF_R;
  float* ta   = w + OFF_TA;
  float* tb   = w + OFF_TB;
  float* part = w + OFF_PART;
  float* coef = w + OFF_COEF;
  _Float16* hf  = (_Float16*)ta;      // reused after diffusion
  _Float16* w1f = (_Float16*)tb;      // reused after diffusion
  _Float16* w2f = w1f + 4 * 4 * 32 * 16;

  dim3 b256(256);
  // symmetric normalization
  k_zero<<<(N + 255) / 256, b256, 0, stream>>>(deg, N);
  k_deg <<<(E + 255) / 256, b256, 0, stream>>>(ei, ew, deg);
  k_dis <<<(N + 255) / 256, b256, 0, stream>>>(deg);
  k_norm<<<(E + 255) / 256, b256, 0, stream>>>(ei, ew, deg, nrm);
  // input projection
  k_proj<<<NH / 256, b256, 0, stream>>>(x, Win, bin, h);

  for (int l = 0; l < L; ++l) {
    k_stats <<<NBLK, 64, 0, stream>>>(h, part);
    k_coeffs<<<1, 64, 0, stream>>>(part, cW1, cb1, cW2, cb2, l, coef);
    k_scale <<<NH / 256, b256, 0, stream>>>(h, R, coef);
    const float* src = h;
    float* dst = ta;
    float* alt = tb;
    for (int k = 1; k <= P; ++k) {
      k_zero<<<NH / 256, b256, 0, stream>>>(dst, NH);
      k_spmm<<<E / 4, b256, 0, stream>>>(ei, nrm, src, dst);
      k_axpy<<<NH / 256, b256, 0, stream>>>(R, dst, coef, k);
      src = dst; float* tmp = dst; dst = alt; alt = tmp;
    }
    k_ln<<<N / 4, b256, 0, stream>>>(h, R, lns, lnb, l);
  }

  // edge predictor (WMMA f16, f32 accum)
  k_h2f16<<<NH / 256, b256, 0, stream>>>(h, hf);
  k_packw<<<40, b256, 0, stream>>>(eW1, eW2, w1f, w2f);
  k_edge <<<E / 128, b256, 0, stream>>>(hf, w1f, w2f, ei, eb1, eb2, eW3, eb3,
                                        (float*)d_out);
  // second output: final h
  k_copy<<<NH / 256, b256, 0, stream>>>(h, (float*)d_out + E);
}
